// EarthAttention3D_64828236366111
// MI455X (gfx1250) — matrix operand order repaired and checked by reading, not compiled, not run on hardware
//
#include <hip/hip_runtime.h>
#include <hip/hip_bf16.h>
#include <math.h>

// ---------------- static problem config (matches reference) ----------------
#define B_      1024          // windows
#define NTOK    144           // tokens per window
#define DIM     192
#define HEADS   6
#define HD      32
#define WT      64            // window types
#define SCALE   0.17677669529663689f   // 32^-0.5
// workspace: q (b,h,n,d) ; k transposed (b,h,d,n) ; v (b,h,n,d)  -- bf16
#define QKV_ELEMS (B_*HEADS*NTOK*HD)   // 28,311,552 per tensor

typedef __attribute__((ext_vector_type(16))) __bf16 v16bf;
typedef __attribute__((ext_vector_type(8)))  float  v8f;

// WMMA A-fragment element -> K offset (16-bit A 16x32 layout, ISA 7.12.2)
//   lanes 0-15 hold K {0..7,16..23}; lanes 16-31 hold K {8..15,24..31}
__device__ __forceinline__ int a_k(int lane, int e) {
  return ((lane < 16) ? 0 : 8) + e + ((e >= 8) ? 8 : 0);
}
// WMMA B-fragment element -> K offset (16-bit B 32x16 layout):
//   lanes 0-15 hold K {0..7,16..23}; lanes 16-31 hold K {8..15,24..31}
__device__ __forceinline__ int b_k(int lane, int e) {
  return ((lane < 16) ? 0 : 8) + e + ((e >= 8) ? 8 : 0);
}

__device__ __forceinline__ v8f wmma_bf16(v16bf a, v16bf b, v8f c) {
  return __builtin_amdgcn_wmma_f32_16x16x32_bf16(false, a, false, b,
                                                 (short)0, c, false, false);
}

// ============================================================================
// Kernel 1: fused QKV projection.  y = x @ qkv_w^T + qkv_b, scattered into
// q (scaled), k (transposed), v workspace as bf16.
// grid = (9 M-tiles, 1024 windows), block = 256 (8 waves). A-frags cached in
// VGPRs across all 36 N-tiles.
// ============================================================================
__global__ void __launch_bounds__(256)
ea3d_qkv_kernel(const float* __restrict__ x,
                const float* __restrict__ qkv_w,
                const float* __restrict__ qkv_b,
                __bf16* __restrict__ q_ws,
                __bf16* __restrict__ k_ws,
                __bf16* __restrict__ v_ws) {
  const int lane = threadIdx.x & 31;
  const int wave = threadIdx.x >> 5;
  const int mt   = blockIdx.x;          // 0..8
  const int b    = blockIdx.y;          // 0..1023

  __builtin_prefetch(qkv_w, 0, 3);      // global_prefetch_b8 (weights -> L2)

  // cache A fragments: 16 rows of x (this M tile), full K=192 -> 6 frags
  const int m = mt * 16 + (lane & 15);
  const float* xrow = x + ((size_t)b * NTOK + m) * DIM;
  v16bf afrag[6];
#pragma unroll
  for (int kk = 0; kk < 6; ++kk)
#pragma unroll
    for (int e = 0; e < 16; ++e)
      afrag[kk][e] = (__bf16)xrow[kk * 32 + a_k(lane, e)];

  for (int nt = wave; nt < 36; nt += 8) {
    const int j = nt * 16 + (lane & 15);       // output column 0..575
    const float* wrow = qkv_w + (size_t)j * DIM;
    v8f c = {};
#pragma unroll
    for (int kk = 0; kk < 6; ++kk) {
      v16bf bfrag;
#pragma unroll
      for (int e = 0; e < 16; ++e)
        bfrag[e] = (__bf16)wrow[kk * 32 + b_k(lane, e)];
      c = wmma_bf16(afrag[kk], bfrag, c);
    }
    const float bias = qkv_b[j];
#pragma unroll
    for (int r = 0; r < 8; ++r) {
      const int mm  = mt * 16 + r + ((lane >> 4) << 3);   // token row
      const float v = c[r] + bias;
      if (j < DIM) {                     // Q, pre-scaled
        const int h = j >> 5, d = j & 31;
        q_ws[(((size_t)b * HEADS + h) * NTOK + mm) * HD + d] = (__bf16)(v * SCALE);
      } else if (j < 2 * DIM) {          // K, transposed [b][h][d][n]
        const int j2 = j - DIM, h = j2 >> 5, d = j2 & 31;
        k_ws[(((size_t)b * HEADS + h) * HD + d) * NTOK + mm] = (__bf16)v;
      } else {                           // V
        const int j3 = j - 2 * DIM, h = j3 >> 5, d = j3 & 31;
        v_ws[(((size_t)b * HEADS + h) * NTOK + mm) * HD + d] = (__bf16)v;
      }
    }
  }
}

// ============================================================================
// Kernel 2: attention core per (window b, head h, 48-row block rb).
//   S = Q@K^T + bias[pos_index] + mask  -> softmax -> O = P@V  (WMMA both)
// K^T (32x144) and V (144x32, zero-padded to 160) are staged once into LDS
// with coalesced 32-bit copies; fragments gather from LDS (64-bank) instead
// of strided global u16 loads. Writes O (f32) directly into d_out.
// grid = (3, 6, 1024), block = 256. LDS = 62 KB.
// ============================================================================
__global__ void __launch_bounds__(256)
ea3d_attn_kernel(const __bf16* __restrict__ q_ws,
                 const __bf16* __restrict__ k_ws,
                 const __bf16* __restrict__ v_ws,
                 const float* __restrict__ bias_table,
                 const int*   __restrict__ pos_idx,
                 const float* __restrict__ mask,
                 float* __restrict__ out) {
  __shared__ float  S[48][144];          // 27648 B
  __shared__ __bf16 P[48][160];          // 15360 B, probs zero-padded K->160
  __shared__ __bf16 Kt[32][160];         // 10240 B, K^T staged (rows: head dim)
  __shared__ __bf16 Vt[160][32];         // 10240 B, V staged, rows 144..159 zero

  const int tid  = threadIdx.x;
  const int lane = tid & 31;
  const int wave = tid >> 5;
  const int rb = blockIdx.x;             // row block 0..2
  const int h  = blockIdx.y;             // head
  const int b  = blockIdx.z;             // window
  const int wt = b & (WT - 1);           // earth window type
  const size_t bh = (size_t)b * HEADS + h;

  __builtin_prefetch(bias_table, 0, 3);  // gather tables live in L2

  // ---- stage K^T and V into LDS (coalesced u32 copies, no conversion) ----
  {
    const uint32_t* ksrc = (const uint32_t*)(k_ws + bh * HD * NTOK);   // 2304 u32
    for (int i = tid; i < 2304; i += 256) {
      const int row = i / 72, w = i % 72;          // 72 u32 per 144-elem row
      *(uint32_t*)&Kt[row][w * 2] = ksrc[i];
    }
    const uint32_t* vsrc = (const uint32_t*)(v_ws + bh * NTOK * HD);   // 2304 u32
    for (int i = tid; i < 2304; i += 256) {
      const int row = i / 16, w = i % 16;          // 16 u32 per 32-elem row
      *(uint32_t*)&Vt[row][w * 2] = vsrc[i];
    }
    for (int i = tid; i < 256; i += 256)           // zero-pad V rows 144..159
      *(uint32_t*)&Vt[144 + i / 16][(i % 16) * 2] = 0u;
  }
  __syncthreads();

  // ---- phase 1: S tiles (3 x 9 tiles, one 16x16x32 WMMA each) ----
  for (int t = wave; t < 27; t += 8) {
    const int mt = t / 9, nt = t % 9;
    v16bf a, kb;
    const int m = rb * 48 + mt * 16 + (lane & 15);
    const __bf16* qrow = q_ws + (bh * NTOK + m) * HD;
#pragma unroll
    for (int e = 0; e < 16; ++e) a[e] = qrow[a_k(lane, e)];
    const int col = nt * 16 + (lane & 15);
#pragma unroll
    for (int e = 0; e < 16; ++e) kb[e] = Kt[b_k(lane, e)][col];
    v8f c = {};
    c = wmma_bf16(a, kb, c);
#pragma unroll
    for (int r = 0; r < 8; ++r) {
      const int ml = mt * 16 + r + ((lane >> 4) << 3);  // local row 0..47
      const int i  = rb * 48 + ml;                      // global row
      const int j  = nt * 16 + (lane & 15);
      const int pi = pos_idx[i * NTOK + j];
      const float bias = bias_table[((size_t)pi * WT + wt) * HEADS + h];
      const float mk   = mask[((size_t)wt * NTOK + i) * NTOK + j];
      S[ml][j] = c[r] + bias + mk;
    }
  }
  __syncthreads();

  // ---- phase 2: softmax, one wave per row (6 rows per wave) ----
#pragma unroll
  for (int rr = 0; rr < 6; ++rr) {
    const int ml = wave * 6 + rr;
    float mx = -INFINITY;
    for (int chn = lane; chn < 144; chn += 32) mx = fmaxf(mx, S[ml][chn]);
#pragma unroll
    for (int off = 16; off; off >>= 1) mx = fmaxf(mx, __shfl_xor(mx, off, 32));
    float sum = 0.f;
    for (int chn = lane; chn < 144; chn += 32) {
      const float e = __expf(S[ml][chn] - mx);
      S[ml][chn] = e;
      sum += e;
    }
#pragma unroll
    for (int off = 16; off; off >>= 1) sum += __shfl_xor(sum, off, 32);
    const float inv = 1.0f / sum;
    for (int chn = lane; chn < 160; chn += 32)
      P[ml][chn] = (chn < 144) ? (__bf16)(S[ml][chn] * inv) : (__bf16)0.0f;
  }
  __syncthreads();

  // ---- phase 3: O = P @ V  (3 M-tiles x 2 N-tiles, K padded to 160) ----
  if (wave < 6) {
    const int mt = wave >> 1, ntO = wave & 1;
    v8f c = {};
#pragma unroll
    for (int kk = 0; kk < 5; ++kk) {
      v16bf a, vb;
      const int ml = mt * 16 + (lane & 15);
#pragma unroll
      for (int e = 0; e < 16; ++e) a[e] = P[ml][kk * 32 + a_k(lane, e)];
      const int d = ntO * 16 + (lane & 15);
#pragma unroll
      for (int e = 0; e < 16; ++e) vb[e] = Vt[kk * 32 + b_k(lane, e)][d];
      c = wmma_bf16(a, vb, c);
    }
#pragma unroll
    for (int r = 0; r < 8; ++r) {
      const int tok = rb * 48 + mt * 16 + r + ((lane >> 4) << 3);
      const int chn = h * HD + ntO * 16 + (lane & 15);
      out[((size_t)b * NTOK + tok) * DIM + chn] = c[r];
    }
  }
}

// ============================================================================
// Kernel 3: output projection, IN PLACE on d_out.
// Each wave owns one 16-row M-tile: loads its full rows into bf16 A-fragment
// registers first, then computes and overwrites all 12 N-tiles (race-free).
// grid = 1152, block = 256 (8 waves -> 9216 M-tiles). Zero LDS.
// ============================================================================
__global__ void __launch_bounds__(256)
ea3d_proj_kernel(float* __restrict__ io,
                 const float* __restrict__ proj_w,
                 const float* __restrict__ proj_b) {
  const int lane = threadIdx.x & 31;
  const int wave = threadIdx.x >> 5;
  const int mt = blockIdx.x * 8 + wave;          // 0..9215
  const int m  = mt * 16 + (lane & 15);

  __builtin_prefetch(proj_w, 0, 3);              // global_prefetch_b8 (L2)

  const float* row = io + (size_t)m * DIM;       // attention output row
  v16bf afrag[6];
#pragma unroll
  for (int kk = 0; kk < 6; ++kk)
#pragma unroll
    for (int e = 0; e < 16; ++e)
      afrag[kk][e] = (__bf16)row[kk * 32 + a_k(lane, e)];

#pragma unroll
  for (int nt = 0; nt < 12; ++nt) {
    const int j = nt * 16 + (lane & 15);
    const float* wrow = proj_w + (size_t)j * DIM;
    v8f c = {};
#pragma unroll
    for (int kk = 0; kk < 6; ++kk) {
      v16bf bfrag;
#pragma unroll
      for (int e = 0; e < 16; ++e)
        bfrag[e] = (__bf16)wrow[kk * 32 + b_k(lane, e)];
      c = wmma_bf16(afrag[kk], bfrag, c);
    }
    const float pb = proj_b[j];
#pragma unroll
    for (int r = 0; r < 8; ++r) {
      const int mm = mt * 16 + r + ((lane >> 4) << 3);
      io[(size_t)mm * DIM + j] = c[r] + pb;
    }
  }
}

// ============================================================================
extern "C" void kernel_launch(void* const* d_in, const int* in_sizes, int n_in,
                              void* d_out, int out_size, void* d_ws, size_t ws_size,
                              hipStream_t stream) {
  const float* x          = (const float*)d_in[0];
  const float* mask       = (const float*)d_in[1];
  const float* qkv_w      = (const float*)d_in[2];
  const float* qkv_b      = (const float*)d_in[3];
  const float* proj_w     = (const float*)d_in[4];
  const float* proj_b     = (const float*)d_in[5];
  const float* bias_table = (const float*)d_in[6];
  const int*   pos_idx    = (const int*)d_in[7];
  float* out = (float*)d_out;

  __bf16* q_ws = (__bf16*)d_ws;                 // needs 3*QKV_ELEMS*2 = 170 MB
  __bf16* k_ws = q_ws + (size_t)QKV_ELEMS;
  __bf16* v_ws = k_ws + (size_t)QKV_ELEMS;

  ea3d_qkv_kernel<<<dim3(9, B_), 256, 0, stream>>>(x, qkv_w, qkv_b,
                                                   q_ws, k_ws, v_ws);
  ea3d_attn_kernel<<<dim3(3, HEADS, B_), 256, 0, stream>>>(q_ws, k_ws, v_ws,
                                                           bias_table, pos_idx,
                                                           mask, out);
  ea3d_proj_kernel<<<dim3(1152), 256, 0, stream>>>(out, proj_w, proj_b);
}
